// TransformerEncoder_8881992368675
// MI455X (gfx1250) — matrix operand order repaired and checked by reading, not compiled, not run on hardware
//
#include <hip/hip_runtime.h>
#include <hip/hip_bf16.h>
#include <cstdint>

// ---------------- problem constants (match reference) ----------------
#define LYR 6
#define DMODEL 768
#define NHEAD 12
#define FFDIM 3072
#define BATCH 2
#define SEQ 2048
#define HDIM 64
#define NTOK (BATCH * SEQ) // 4096

typedef __bf16 bf16;
typedef bf16  bf16x8  __attribute__((ext_vector_type(8)));
typedef bf16  bf16x16 __attribute__((ext_vector_type(16)));
typedef float floatx8 __attribute__((ext_vector_type(8)));

__device__ __forceinline__ bf16x16 cat8(bf16x8 lo, bf16x8 hi) {
    return __builtin_shufflevector(lo, hi, 0, 1, 2, 3, 4, 5, 6, 7,
                                   8, 9, 10, 11, 12, 13, 14, 15);
}

// ---------------------------------------------------------------------
// Tensor Data Mover path (gfx1250 only, guarded): one wave issues 2D
// tile DMA descriptors; TENSORcnt tracks completion (in-order per wave,
// which the double-buffer pipeline below relies on). Falls back to
// cooperative global->LDS staging when the builtin is unavailable.
// ---------------------------------------------------------------------
#if defined(__gfx1250__) && __has_builtin(__builtin_amdgcn_tensor_load_to_lds) && \
    __has_builtin(__builtin_amdgcn_s_wait_tensorcnt)
#define USE_TDM 1
#endif

#ifdef USE_TDM
typedef unsigned int u32x4 __attribute__((ext_vector_type(4)));
typedef int          i32x4 __attribute__((ext_vector_type(4)));
typedef int          i32x8 __attribute__((ext_vector_type(8)));

// 2D tile load: tile_d0 elements along contiguous dim (K), tile_d1 rows.
// Lands row-major in LDS: row r at lds_off + r*tile_d0*2 bytes, which is
// exactly the sA/sB layout the WMMA fragment reads use.
__device__ __forceinline__ void tdm_load_2d(unsigned lds_off, const bf16* gaddr,
                                            unsigned tensor_d0, unsigned tensor_d1,
                                            unsigned tile_d0, unsigned tile_d1,
                                            unsigned long long stride0_elems) {
    unsigned long long ga = (unsigned long long)(uintptr_t)gaddr;
    u32x4 g0;
    g0[0] = 1u;                                      // count=1 valid descriptor
    g0[1] = lds_off;                                 // lds_addr [63:32]
    g0[2] = (unsigned)(ga & 0xFFFFFFFFu);            // global_addr lo
    g0[3] = (unsigned)((ga >> 32) & 0x01FFFFFFu)     // global_addr hi [120:96]
          | 0x80000000u;                             // type=2 ("image") [127:126]
    i32x8 g1;
    g1[0] = (int)(1u << 16);                         // data_size=1 -> 2 bytes/elem
    g1[1] = (int)((tensor_d0 & 0xFFFFu) << 16);      // tensor_dim0 [79:48] lo
    g1[2] = (int)(((tensor_d0 >> 16) & 0xFFFFu)      // tensor_dim0 hi
          | ((tensor_d1 & 0xFFFFu) << 16));          // tensor_dim1 [111:80] lo
    g1[3] = (int)(((tensor_d1 >> 16) & 0xFFFFu)      // tensor_dim1 hi
          | ((tile_d0 & 0xFFFFu) << 16));            // tile_dim0 [127:112]
    g1[4] = (int)(tile_d1 & 0xFFFFu);                // tile_dim1; tile_dim2=0
    g1[5] = (int)(stride0_elems & 0xFFFFFFFFull);    // tensor_dim0_stride lo
    g1[6] = (int)((stride0_elems >> 32) & 0xFFFFull);// stride0 hi; stride1=0
    g1[7] = 0;
    i32x4 z4 = {0, 0, 0, 0};
#if __clang_major__ >= 23
    i32x8 z8 = {0, 0, 0, 0, 0, 0, 0, 0};
    __builtin_amdgcn_tensor_load_to_lds(g0, g1, z4, z4, z8, 0);
#else
    __builtin_amdgcn_tensor_load_to_lds(g0, g1, z4, z4, 0);
#endif
}
#endif // USE_TDM

// ---------------------------------------------------------------------
// Weight cast + transpose: W[K][N] fp32 (d_in, d_out) -> Wt[N][K] bf16
// so GEMM B-fragments are contiguous-K b128 loads.
// ---------------------------------------------------------------------
__global__ void castT_kernel(const float* __restrict__ W, bf16* __restrict__ Wt,
                             int Kd, int Nd) {
    size_t idx = (size_t)blockIdx.x * 256 + threadIdx.x;
    if (idx >= (size_t)Kd * Nd) return;
    int kk = (int)(idx / Nd);
    int nn = (int)(idx % Nd);
    Wt[(size_t)nn * Kd + kk] = (bf16)W[idx];
}

// ---------------------------------------------------------------------
// LayerNorm over D=768 (one 256-thread block per token row).
// Emits fp32 (residual stream) and/or bf16 (GEMM A input).
// ---------------------------------------------------------------------
__global__ __launch_bounds__(256) void ln_kernel(const float* __restrict__ x,
                                                 const float* __restrict__ w,
                                                 const float* __restrict__ b,
                                                 float* __restrict__ outF,
                                                 bf16* __restrict__ outBF) {
    const int row = blockIdx.x;
    const int t   = threadIdx.x;
    const float* xr = x + (size_t)row * DMODEL;
    float v0 = xr[t], v1 = xr[t + 256], v2 = xr[t + 512];

    __shared__ float red[8];
    float s = v0 + v1 + v2;
    for (int m = 1; m < 32; m <<= 1) s += __shfl_xor(s, m, 32);
    if ((t & 31) == 0) red[t >> 5] = s;
    __syncthreads();
    float tot = 0.f;
    for (int i = 0; i < 8; i++) tot += red[i];
    float mu = tot * (1.0f / DMODEL);

    float d0 = v0 - mu, d1 = v1 - mu, d2 = v2 - mu;
    float sq = d0 * d0 + d1 * d1 + d2 * d2;
    __syncthreads();
    for (int m = 1; m < 32; m <<= 1) sq += __shfl_xor(sq, m, 32);
    if ((t & 31) == 0) red[t >> 5] = sq;
    __syncthreads();
    float tv = 0.f;
    for (int i = 0; i < 8; i++) tv += red[i];
    float rstd = rsqrtf(tv * (1.0f / DMODEL) + 1e-5f);

    float y0 = d0 * rstd * w[t]       + b[t];
    float y1 = d1 * rstd * w[t + 256] + b[t + 256];
    float y2 = d2 * rstd * w[t + 512] + b[t + 512];
    size_t base = (size_t)row * DMODEL;
    if (outF) {
        outF[base + t] = y0; outF[base + t + 256] = y1; outF[base + t + 512] = y2;
    }
    if (outBF) {
        outBF[base + t] = (bf16)y0; outBF[base + t + 256] = (bf16)y1;
        outBF[base + t + 512] = (bf16)y2;
    }
}

// ---------------------------------------------------------------------
// bf16 WMMA GEMM: out[M][N] = A[M][K] @ Bt[N][K]^T + bias (+res)(+gelu)
// block = 256 threads = 8 waves; block tile 128x64; wave tile 32x32
// (2x2 v_wmma_f32_16x16x32_bf16 frags), K stepped by 32.
// TDM mode: double-buffered LDS; wave 0 issues the NEXT tile pair before
// consuming the current one, then s_wait_tensorcnt(2) — TDM completion is
// in-order per wave, so <=2 outstanding proves the older pair landed.
// ---------------------------------------------------------------------
#define SA_TILE (128 * 32)
#define SB_TILE (64 * 32)

__global__ __launch_bounds__(256) void gemm_bf16_kernel(
    const bf16* __restrict__ A, const bf16* __restrict__ Bt,
    const float* __restrict__ bias, const float* __restrict__ res,
    float* __restrict__ outF, bf16* __restrict__ outBF,
    int M, int Nn, int Kn, int act_gelu) {
#ifdef USE_TDM
    __shared__ __attribute__((aligned(16))) bf16 sA[2 * SA_TILE];
    __shared__ __attribute__((aligned(16))) bf16 sB[2 * SB_TILE];
#else
    __shared__ __attribute__((aligned(16))) bf16 sA[SA_TILE];
    __shared__ __attribute__((aligned(16))) bf16 sB[SB_TILE];
#endif

    const int m0 = blockIdx.x * 128;
    const int n0 = blockIdx.y * 64;
    const int t    = threadIdx.x;
    const int lane = t & 31, wave = t >> 5;
    const int wm = (wave >> 1) * 32, wn = (wave & 1) * 32;
    const int lr = lane & 15, lh = lane >> 4;

    floatx8 acc[2][2];
    for (int i = 0; i < 2; i++)
        for (int j = 0; j < 2; j++)
            for (int r = 0; r < 8; r++) acc[i][j][r] = 0.f;

#ifdef USE_TDM
    const int ksteps = Kn / 32;
    if (wave == 0) { // prologue: DMA first tile pair into buffer 0
        tdm_load_2d((unsigned)(uintptr_t)&sA[0], A + (size_t)m0 * Kn,
                    (unsigned)Kn, (unsigned)M, 32u, 128u, (unsigned long long)Kn);
        tdm_load_2d((unsigned)(uintptr_t)&sB[0], Bt + (size_t)n0 * Kn,
                    (unsigned)Kn, (unsigned)Nn, 32u, 64u, (unsigned long long)Kn);
    }
    for (int ks = 0; ks < ksteps; ks++) {
        const int cur = ks & 1, nxt = cur ^ 1;
        if (wave == 0) {
            if (ks + 1 < ksteps) { // pipeline next tile pair under compute
                const int k1 = (ks + 1) * 32;
                tdm_load_2d((unsigned)(uintptr_t)&sA[nxt * SA_TILE],
                            A + (size_t)m0 * Kn + k1,
                            (unsigned)Kn, (unsigned)M, 32u, 128u,
                            (unsigned long long)Kn);
                tdm_load_2d((unsigned)(uintptr_t)&sB[nxt * SB_TILE],
                            Bt + (size_t)n0 * Kn + k1,
                            (unsigned)Kn, (unsigned)Nn, 32u, 64u,
                            (unsigned long long)Kn);
                __builtin_amdgcn_s_wait_tensorcnt(2); // oldest pair (cur) done
            } else {
                __builtin_amdgcn_s_wait_tensorcnt(0);
            }
            asm volatile("" ::: "memory");
        }
        __syncthreads(); // cur buffer ready for all waves
        const bf16* cA = &sA[cur * SA_TILE];
        const bf16* cB = &sB[cur * SB_TILE];

        bf16x16 af[2], bfm[2];
        for (int mi = 0; mi < 2; mi++) {
            int row = wm + mi * 16 + lr, kb = lh * 8;
            af[mi] = cat8(*(const bf16x8*)&cA[row * 32 + kb],
                          *(const bf16x8*)&cA[row * 32 + 16 + kb]);
        }
        for (int ni = 0; ni < 2; ni++) {
            int col = wn + ni * 16 + lr, kb2 = lh * 8;
            bfm[ni] = cat8(*(const bf16x8*)&cB[col * 32 + kb2],
                           *(const bf16x8*)&cB[col * 32 + 16 + kb2]);
        }
        for (int mi = 0; mi < 2; mi++)
            for (int ni = 0; ni < 2; ni++)
                acc[mi][ni] = __builtin_amdgcn_wmma_f32_16x16x32_bf16(
                    false, af[mi], false, bfm[ni], (short)0, acc[mi][ni],
                    false, false);
        __syncthreads(); // all waves done reading cur before it is re-DMA'd
    }
#else
    // ---- fallback: cooperative staging (b128 global loads -> ds stores)
    const int arow = t >> 1, ahalf = t & 1; // A tile: 128 rows x 32 K
    const int brow = t >> 2, bseg = t & 3;  // B tile: 64 cols x 32 K
    for (int k0 = 0; k0 < Kn; k0 += 32) {
        {
            const bf16* src = A + (size_t)(m0 + arow) * Kn + k0 + ahalf * 16;
            *(bf16x8*)&sA[arow * 32 + ahalf * 16]     = *(const bf16x8*)src;
            *(bf16x8*)&sA[arow * 32 + ahalf * 16 + 8] = *(const bf16x8*)(src + 8);
            const bf16* srcB = Bt + (size_t)(n0 + brow) * Kn + k0 + bseg * 8;
            *(bf16x8*)&sB[brow * 32 + bseg * 8] = *(const bf16x8*)srcB;
        }
        if (k0 + 32 < Kn) {
            __builtin_prefetch(A  + (size_t)(m0 + arow) * Kn + k0 + 32 + ahalf * 16, 0, 1);
            __builtin_prefetch(Bt + (size_t)(n0 + brow) * Kn + k0 + 32 + bseg * 8, 0, 1);
        }
        __syncthreads();
        bf16x16 af[2], bfm[2];
        for (int mi = 0; mi < 2; mi++) {
            int row = wm + mi * 16 + lr, kb = lh * 8;
            af[mi] = cat8(*(const bf16x8*)&sA[row * 32 + kb],
                          *(const bf16x8*)&sA[row * 32 + 16 + kb]);
        }
        for (int ni = 0; ni < 2; ni++) {
            int col = wn + ni * 16 + lr, kb2 = lh * 8;
            bfm[ni] = cat8(*(const bf16x8*)&sB[col * 32 + kb2],
                           *(const bf16x8*)&sB[col * 32 + 16 + kb2]);
        }
        for (int mi = 0; mi < 2; mi++)
            for (int ni = 0; ni < 2; ni++)
                acc[mi][ni] = __builtin_amdgcn_wmma_f32_16x16x32_bf16(
                    false, af[mi], false, bfm[ni], (short)0, acc[mi][ni],
                    false, false);
        __syncthreads();
    }
#endif

    // ---- fused epilogue: bias + residual + GELU + fp32/bf16 stores
    for (int mi = 0; mi < 2; mi++)
        for (int ni = 0; ni < 2; ni++)
            for (int r = 0; r < 8; r++) {
                int row = m0 + wm + mi * 16 + lh * 8 + r;
                int col = n0 + wn + ni * 16 + lr;
                float v = acc[mi][ni][r] + bias[col];
                size_t idx = (size_t)row * Nn + col;
                if (res) v += res[idx];
                if (act_gelu) v = 0.5f * v * (1.0f + erff(v * 0.70710678118654752f));
                if (outF)  outF[idx]  = v;
                if (outBF) outBF[idx] = (bf16)v;
            }
}

// ---------------------------------------------------------------------
// Flash-style attention: one wave per 16-query tile of one (b,h).
// Scores via 2x v_wmma (K-dim 64 = 2x32); online softmax with
// width-16 shfl row reductions + v_exp_f32; P transposed C->A layout
// through a per-wave LDS slab; P@V via WMMA (keys 16..31 zero-padded).
// ---------------------------------------------------------------------
__global__ __launch_bounds__(256) void attn_kernel(const bf16* __restrict__ q,
                                                   const bf16* __restrict__ k,
                                                   const bf16* __restrict__ v,
                                                   bf16* __restrict__ o_out) {
    __shared__ bf16 lds_p[8 * 256]; // 16x16 bf16 P tile per wave
    const int bh = blockIdx.x;
    const int b = bh / NHEAD, h = bh % NHEAD;
    const int lane = threadIdx.x & 31, wave = threadIdx.x >> 5;
    const int lr = lane & 15, lh = lane >> 4;
    const int q0 = (blockIdx.y * 8 + wave) * 16;
    const size_t base = ((size_t)b * SEQ) * DMODEL + (size_t)h * HDIM;

    // Q fragments (A layout), dims 0..31 and 32..63
    bf16x16 qf0, qf1;
    {
        size_t g = base + (size_t)(q0 + lr) * DMODEL;
        int kb = lh * 8;
        qf0 = cat8(*(const bf16x8*)(q + g + kb),      *(const bf16x8*)(q + g + 16 + kb));
        qf1 = cat8(*(const bf16x8*)(q + g + 32 + kb), *(const bf16x8*)(q + g + 48 + kb));
    }

    float m_[8], l_[8];
    floatx8 acc[4];
    for (int r = 0; r < 8; r++) { m_[r] = -1e30f; l_[r] = 0.f; }
    for (int tI = 0; tI < 4; tI++)
        for (int r = 0; r < 8; r++) acc[tI][r] = 0.f;
    floatx8 zero8;
    for (int r = 0; r < 8; r++) zero8[r] = 0.f;

    for (int kt = 0; kt < SEQ / 16; kt++) {
        const int key0 = kt * 16;
        // K fragments in B layout straight from row-major K (table order, matches Q)
        size_t gk = base + (size_t)(key0 + lr) * DMODEL;
        int kb2 = lh * 8;
        bf16x16 kf0 = cat8(*(const bf16x8*)(k + gk + kb2),
                           *(const bf16x8*)(k + gk + 16 + kb2));
        bf16x16 kf1 = cat8(*(const bf16x8*)(k + gk + 32 + kb2),
                           *(const bf16x8*)(k + gk + 48 + kb2));
        floatx8 sc = __builtin_amdgcn_wmma_f32_16x16x32_bf16(
            false, qf0, false, kf0, (short)0, zero8, false, false);
        sc = __builtin_amdgcn_wmma_f32_16x16x32_bf16(
            false, qf1, false, kf1, (short)0, sc, false, false);

        // online softmax per row (rows live across 16-lane halves)
        for (int r = 0; r < 8; r++) {
            float sv = sc[r] * 0.125f; // 1/sqrt(64)
            float mx = sv;
            for (int msk = 1; msk < 16; msk <<= 1)
                mx = fmaxf(mx, __shfl_xor(mx, msk, 16));
            float mnew = fmaxf(m_[r], mx);
            float p = __expf(sv - mnew);
            float rs = p;
            for (int msk = 1; msk < 16; msk <<= 1) rs += __shfl_xor(rs, msk, 16);
            float alpha = __expf(m_[r] - mnew);
            l_[r] = l_[r] * alpha + rs;
            m_[r] = mnew;
            for (int tI = 0; tI < 4; tI++) acc[tI][r] *= alpha;
            // C-layout -> LDS (row = query, col = key)
            lds_p[wave * 256 + (lh * 8 + r) * 16 + lr] = (bf16)p;
        }
        // read back in A layout (per-wave slab; same-wave DS ops are in order)
        bf16x16 pf;
        for (int i = 0; i < 8; i++)
            pf[i] = lds_p[wave * 256 + lr * 16 + lh * 8 + i];
        for (int i = 8; i < 16; i++) pf[i] = (bf16)0.0f; // keys 16..31 zero

        for (int tI = 0; tI < 4; tI++) {
            bf16x16 vf;
            for (int i = 0; i < 16; i++)
                vf[i] = (i < 8)
                            ? v[base + (size_t)(key0 + lh * 8 + i) * DMODEL + tI * 16 + lr]
                            : (bf16)0.0f;
            acc[tI] = __builtin_amdgcn_wmma_f32_16x16x32_bf16(
                false, pf, false, vf, (short)0, acc[tI], false, false);
        }
    }

    for (int tI = 0; tI < 4; tI++)
        for (int r = 0; r < 8; r++) {
            size_t g = base + (size_t)(q0 + lh * 8 + r) * DMODEL + tI * 16 + lr;
            o_out[g] = (bf16)(acc[tI][r] / l_[r]);
        }
}

// ---------------------------------------------------------------------
extern "C" void kernel_launch(void* const* d_in, const int* in_sizes, int n_in,
                              void* d_out, int out_size, void* d_ws, size_t ws_size,
                              hipStream_t stream) {
    const float* X    = (const float*)d_in[0];
    const float* Wq   = (const float*)d_in[1];  const float* bq   = (const float*)d_in[2];
    const float* Wk   = (const float*)d_in[3];  const float* bk   = (const float*)d_in[4];
    const float* Wv   = (const float*)d_in[5];  const float* bv   = (const float*)d_in[6];
    const float* Wo   = (const float*)d_in[7];  const float* bo   = (const float*)d_in[8];
    const float* ln1w = (const float*)d_in[9];  const float* ln1b = (const float*)d_in[10];
    const float* W1   = (const float*)d_in[11]; const float* b1   = (const float*)d_in[12];
    const float* W2   = (const float*)d_in[13]; const float* b2   = (const float*)d_in[14];
    const float* ln2w = (const float*)d_in[15]; const float* ln2b = (const float*)d_in[16];
    const float* lnfw = (const float*)d_in[17]; const float* lnfb = (const float*)d_in[18];

    char* ws = (char*)d_ws;
    size_t off = 0;
    auto carve = [&](size_t bytes) -> char* {
        off = (off + 255) & ~(size_t)255;
        char* p = ws + off;
        off += bytes;
        return p;
    };
    const size_t DD = (size_t)DMODEL * DMODEL;
    const size_t DF = (size_t)DMODEL * FFDIM;
    bf16* Wqt = (bf16*)carve(LYR * DD * 2);
    bf16* Wkt = (bf16*)carve(LYR * DD * 2);
    bf16* Wvt = (bf16*)carve(LYR * DD * 2);
    bf16* Wot = (bf16*)carve(LYR * DD * 2);
    bf16* W1t = (bf16*)carve(LYR * DF * 2); // [F][D]
    bf16* W2t = (bf16*)carve(LYR * DF * 2); // [D][F]
    float* xbuf = (float*)carve((size_t)NTOK * DMODEL * 4); // residual stream
    float* xn   = (float*)carve((size_t)NTOK * DMODEL * 4); // normalized (residual src)
    bf16* xnb   = (bf16*)carve((size_t)NTOK * DMODEL * 2);
    bf16* qB    = (bf16*)carve((size_t)NTOK * DMODEL * 2);
    bf16* kB    = (bf16*)carve((size_t)NTOK * DMODEL * 2);
    bf16* vB    = (bf16*)carve((size_t)NTOK * DMODEL * 2);
    bf16* aB    = (bf16*)carve((size_t)NTOK * DMODEL * 2);
    bf16* hB    = (bf16*)carve((size_t)NTOK * FFDIM * 2);

    hipMemcpyAsync(xbuf, X, (size_t)NTOK * DMODEL * sizeof(float),
                   hipMemcpyDeviceToDevice, stream);

    // one-time weight cast+transpose to bf16 [Nout][K]
    const int nDD = (int)((DD + 255) / 256);
    const int nDF = (int)((DF + 255) / 256);
    for (int i = 0; i < LYR; i++) {
        castT_kernel<<<nDD, 256, 0, stream>>>(Wq + i * DD, Wqt + i * DD, DMODEL, DMODEL);
        castT_kernel<<<nDD, 256, 0, stream>>>(Wk + i * DD, Wkt + i * DD, DMODEL, DMODEL);
        castT_kernel<<<nDD, 256, 0, stream>>>(Wv + i * DD, Wvt + i * DD, DMODEL, DMODEL);
        castT_kernel<<<nDD, 256, 0, stream>>>(Wo + i * DD, Wot + i * DD, DMODEL, DMODEL);
        castT_kernel<<<nDF, 256, 0, stream>>>(W1 + i * DF, W1t + i * DF, DMODEL, FFDIM);
        castT_kernel<<<nDF, 256, 0, stream>>>(W2 + i * DF, W2t + i * DF, FFDIM, DMODEL);
    }

    dim3 gProjD(NTOK / 128, DMODEL / 64);
    dim3 gFF1(NTOK / 128, FFDIM / 64);
    dim3 gAttn(BATCH * NHEAD, SEQ / 128);

    for (int i = 0; i < LYR; i++) {
        ln_kernel<<<NTOK, 256, 0, stream>>>(xbuf, ln1w + (size_t)i * DMODEL,
                                            ln1b + (size_t)i * DMODEL, xn, xnb);
        gemm_bf16_kernel<<<gProjD, 256, 0, stream>>>(xnb, Wqt + i * DD,
            bq + (size_t)i * DMODEL, nullptr, nullptr, qB, NTOK, DMODEL, DMODEL, 0);
        gemm_bf16_kernel<<<gProjD, 256, 0, stream>>>(xnb, Wkt + i * DD,
            bk + (size_t)i * DMODEL, nullptr, nullptr, kB, NTOK, DMODEL, DMODEL, 0);
        gemm_bf16_kernel<<<gProjD, 256, 0, stream>>>(xnb, Wvt + i * DD,
            bv + (size_t)i * DMODEL, nullptr, nullptr, vB, NTOK, DMODEL, DMODEL, 0);
        attn_kernel<<<gAttn, 256, 0, stream>>>(qB, kB, vB, aB);
        gemm_bf16_kernel<<<gProjD, 256, 0, stream>>>(aB, Wot + i * DD,
            bo + (size_t)i * DMODEL, xn, xbuf, nullptr, NTOK, DMODEL, DMODEL, 0);
        ln_kernel<<<NTOK, 256, 0, stream>>>(xbuf, ln2w + (size_t)i * DMODEL,
                                            ln2b + (size_t)i * DMODEL, xn, xnb);
        gemm_bf16_kernel<<<gFF1, 256, 0, stream>>>(xnb, W1t + i * DF,
            b1 + (size_t)i * FFDIM, nullptr, nullptr, hB, NTOK, FFDIM, DMODEL, 1);
        gemm_bf16_kernel<<<gProjD, 256, 0, stream>>>(hB, W2t + i * DF,
            b2 + (size_t)i * DMODEL, xn, xbuf, nullptr, NTOK, DMODEL, FFDIM, 0);
    }
    ln_kernel<<<NTOK, 256, 0, stream>>>(xbuf, lnfw, lnfb, (float*)d_out, nullptr);
}
